// SlotAttentionPerspectiveGenerator_8538394984495
// MI455X (gfx1250) — matrix operand order repaired and checked by reading, not compiled, not run on hardware
//
#include <hip/hip_runtime.h>
#include <hip/hip_bf16.h>

// ---------------------------------------------------------------------------
// Slot Attention on MI455X (gfx1250). All GEMMs via v_wmma_f32_16x16x32_f16
// (wave32, fp32 accumulate), register-blocked so each wave computes multiple
// 16x16 tiles per K-step: the k/v projection does 8 WMMAs per K-step from
// 4 A-fragments + 2 B-fragments (~24 B/lane per WMMA of L2 traffic).
// Weights pre-transposed to [in,out] fp16 so B-fragments are contiguous.
// ---------------------------------------------------------------------------

typedef __attribute__((ext_vector_type(16))) _Float16 v16h;
typedef __attribute__((ext_vector_type(8)))  _Float16 v8h;
typedef __attribute__((ext_vector_type(8)))  float    v8f;

#define B_   32
#define L_   2048
#define D_   1024
#define N_   16
#define H_   2048
#define NIT_ 4
#define EPS_     1e-8f
#define RMS_EPS_ 1e-6f

// ---- WMMA fragment helpers (layouts per CDNA5 ISA 7.12.2) ------------------

// A: 16x32 fp16 tile from row-major [*, lda]; lane&15 = M row,
// lanes 0-15 hold K {0..7, 16..23}, lanes 16-31 hold K {8..15, 24..31}.
static __device__ __forceinline__ v16h load_a16x32(const _Float16* A, int lda,
                                                   int k0, int lane) {
  const _Float16* p = A + (lane & 15) * lda + k0 + ((lane >> 4) << 3);
  v8h lo = *(const v8h*)p;
  v8h hi = *(const v8h*)(p + 16);
  v16h r;
#pragma unroll
  for (int i = 0; i < 8; ++i) { r[i] = lo[i]; r[i + 8] = hi[i]; }
  return r;
}

// B: 32x16 fp16 tile from row-major [K, ldb] (pre-transposed weight [in,out]);
// lane&15 = N column, lanes 0-15 hold K {0..7, 16..23}, lanes 16-31 hold K {8..15, 24..31}.
static __device__ __forceinline__ v16h load_b32x16(const _Float16* Bm, int ldb,
                                                   int k0, int n0, int lane) {
  const _Float16* p = Bm + (size_t)(k0 + ((lane >> 4) << 3)) * ldb + n0 + (lane & 15);
  v16h r;
#pragma unroll
  for (int i = 0; i < 8; ++i) { r[i] = p[(size_t)i * ldb]; r[i + 8] = p[(size_t)(i + 16) * ldb]; }
  return r;
}

static __device__ __forceinline__ v8f wmma_f16(v16h a, v16h b, v8f c) {
  return __builtin_amdgcn_wmma_f32_16x16x32_f16(false, a, false, b,
                                                (short)0, c, false, false);
}

static __device__ __forceinline__ float sigmoidf_(float x) {
  return 1.0f / (1.0f + __expf(-x));
}

// ---- Small prep kernels ----------------------------------------------------

__global__ void k_transpose_f16(const float* __restrict__ src,
                                _Float16* __restrict__ dst, int R, int C) {
  int idx = blockIdx.x * blockDim.x + threadIdx.x;
  if (idx >= R * C) return;
  int r = idx / C, c = idx % C;
  dst[(size_t)c * R + r] = (_Float16)src[(size_t)r * C + c];
}

__global__ void k_slots_init(const float* __restrict__ noise,
                             const float* __restrict__ mu,
                             const float* __restrict__ sigma,
                             float* __restrict__ slots) {
  int idx = blockIdx.x * blockDim.x + threadIdx.x;
  if (idx >= B_ * N_ * D_) return;
  int d = idx % D_;
  slots[idx] = mu[d] + sigma[d] * noise[idx];
}

// rmsnorm one row of D_ fp32 -> fp16  (grid = rows, block = 256)
__global__ void k_rmsnorm_f16(const float* __restrict__ x,
                              const float* __restrict__ g,
                              _Float16* __restrict__ out) {
  int row = blockIdx.x;
  const float* xr = x + (size_t)row * D_;
  float ss = 0.f;
  for (int i = threadIdx.x; i < D_; i += 256) { float v = xr[i]; ss += v * v; }
  __shared__ float sred[8];
  for (int m = 16; m >= 1; m >>= 1) ss += __shfl_xor(ss, m, 32);
  if ((threadIdx.x & 31) == 0) sred[threadIdx.x >> 5] = ss;
  __syncthreads();
  float tot = 0.f;
#pragma unroll
  for (int i = 0; i < 8; ++i) tot += sred[i];
  float rs = rsqrtf(tot * (1.0f / D_) + RMS_EPS_);
  for (int i = threadIdx.x; i < D_; i += 256)
    out[(size_t)row * D_ + i] = (_Float16)(xr[i] * rs * g[i]);
}

// per-iteration: rmsnorm(slots)->s_h, raw slots->fp16, zero attn_sum
__global__ void k_slot_prep(const float* __restrict__ slots,
                            const float* __restrict__ g,
                            _Float16* __restrict__ s_h,
                            _Float16* __restrict__ slots_h,
                            float* __restrict__ attn_sum) {
  int row = blockIdx.x;  // b*N_+n
  const float* xr = slots + (size_t)row * D_;
  float ss = 0.f;
  for (int i = threadIdx.x; i < D_; i += 256) { float v = xr[i]; ss += v * v; }
  __shared__ float sred[8];
  for (int m = 16; m >= 1; m >>= 1) ss += __shfl_xor(ss, m, 32);
  if ((threadIdx.x & 31) == 0) sred[threadIdx.x >> 5] = ss;
  __syncthreads();
  float tot = 0.f;
#pragma unroll
  for (int i = 0; i < 8; ++i) tot += sred[i];
  float rs = rsqrtf(tot * (1.0f / D_) + RMS_EPS_);
  for (int i = threadIdx.x; i < D_; i += 256) {
    float v = xr[i];
    s_h[(size_t)row * D_ + i]     = (_Float16)(v * rs * g[i]);
    slots_h[(size_t)row * D_ + i] = (_Float16)v;
  }
  if (threadIdx.x == 0) attn_sum[row] = 0.f;
}

// ---- WMMA GEMM kernels -----------------------------------------------------

// k = x @ WkT, v = x @ WvT fused; 4 row-tiles per wave share the B-fragments:
// 8 WMMAs per K-step from 4 A + 2 B fragment loads.
__global__ void k_kv_gemm(const _Float16* __restrict__ x,
                          const _Float16* __restrict__ wkT,
                          const _Float16* __restrict__ wvT,
                          _Float16* __restrict__ kx, _Float16* __restrict__ vx) {
  int lane = threadIdx.x;
  int n0 = blockIdx.x * 16;                 // output column (D)
  int r0 = blockIdx.y * 64;                 // 4 row-tiles over B*L
  v8f ck[4] = {};
  v8f cv[4] = {};
  for (int k = 0; k < D_; k += 32) {
    v16h bk = load_b32x16(wkT, D_, k, n0, lane);
    v16h bv = load_b32x16(wvT, D_, k, n0, lane);
#pragma unroll
    for (int t = 0; t < 4; ++t) {
      v16h a = load_a16x32(x + (size_t)(r0 + 16 * t) * D_, D_, k, lane);
      ck[t] = wmma_f16(a, bk, ck[t]);
      cv[t] = wmma_f16(a, bv, cv[t]);
    }
  }
  int col = lane & 15, rb = (lane >> 4) << 3;
#pragma unroll
  for (int t = 0; t < 4; ++t) {
#pragma unroll
    for (int i = 0; i < 8; ++i) {
      size_t o = (size_t)(r0 + 16 * t + rb + i) * D_ + n0 + col;
      kx[o] = (_Float16)ck[t][i];
      vx[o] = (_Float16)cv[t][i];
    }
  }
}

// q = s @ WqT, 4 column-tiles per wave share the A-fragment; stored
// TRANSPOSED as qT[b][d][n] so the attention kernel's B-fragments are
// contiguous.
__global__ void k_q_proj(const _Float16* __restrict__ s_h,
                         const _Float16* __restrict__ wqT,
                         _Float16* __restrict__ qT) {
  int lane = threadIdx.x;
  int n0 = blockIdx.x * 64;                 // 4 d-tiles
  int b = blockIdx.y;
  const _Float16* A = s_h + (size_t)b * N_ * D_;
  v8f c[4] = {};
  for (int k = 0; k < D_; k += 32) {
    v16h a = load_a16x32(A, D_, k, lane);
#pragma unroll
    for (int t = 0; t < 4; ++t) {
      v16h bf = load_b32x16(wqT, D_, k, n0 + 16 * t, lane);
      c[t] = wmma_f16(a, bf, c[t]);
    }
  }
  int dl = lane & 15, rb = (lane >> 4) << 3;
  _Float16* q = qT + (size_t)b * D_ * N_;
#pragma unroll
  for (int t = 0; t < 4; ++t)
#pragma unroll
    for (int i = 0; i < 8; ++i)
      q[(size_t)(n0 + 16 * t + dl) * N_ + rb + i] = (_Float16)c[t][i];
}

// dotsT tiles = k-rows (64xD) @ qT (Dx16); the single qT B-fragment feeds 4
// WMMAs per K-step. Softmax over slots (= the 16 lanes of each wave half) via
// shfl_xor 1/2/4/8; attn stored [N, L] so each lane writes 8 contiguous fp16;
// per-slot row sums accumulated atomically.
__global__ void k_attn(const _Float16* __restrict__ kx,
                       const _Float16* __restrict__ qT,
                       _Float16* __restrict__ attn,
                       float* __restrict__ attn_sum) {
  int lane = threadIdx.x;
  int l0 = blockIdx.x * 64;                 // 4 L-tiles
  int b = blockIdx.y;
  const _Float16* Bq = qT + (size_t)b * D_ * N_;
  v8f c[4] = {};
  for (int k = 0; k < D_; k += 32) {
    v16h bf = load_b32x16(Bq, N_, k, 0, lane);
#pragma unroll
    for (int t = 0; t < 4; ++t) {
      v16h a =
          load_a16x32(kx + ((size_t)b * L_ + l0 + 16 * t) * D_, D_, k, lane);
      c[t] = wmma_f16(a, bf, c[t]);
    }
  }
  const float scale = 0.03125f;             // D^-0.5
  int n = lane & 15, mb = (lane >> 4) << 3;
  float part = 0.f;
#pragma unroll
  for (int t = 0; t < 4; ++t) {
    v8h outv;
#pragma unroll
    for (int i = 0; i < 8; ++i) {
      float x = c[t][i] * scale;
      float mx = x;
      for (int m = 8; m >= 1; m >>= 1) mx = fmaxf(mx, __shfl_xor(mx, m, 32));
      float e = __expf(x - mx);
      float s = e;
      for (int m = 8; m >= 1; m >>= 1) s += __shfl_xor(s, m, 32);
      float a = e / s;                      // softmax over the slot axis
      part += a;
      outv[i] = (_Float16)(a + EPS_);
    }
    *(v8h*)(attn + ((size_t)b * N_ + n) * L_ + l0 + 16 * t + mb) = outv;
  }
  part += __shfl_xor(part, 16, 32);
  if (lane < 16) atomicAdd(attn_sum + b * N_ + n, part);
}

// updates = (attn+eps) @ v, normalized by per-slot attn sum; 4 column-tiles
// per wave share the A-fragment.
__global__ void k_updates(const _Float16* __restrict__ attn,
                          const _Float16* __restrict__ vx,
                          const float* __restrict__ attn_sum,
                          _Float16* __restrict__ u_h) {
  int lane = threadIdx.x;
  int d0 = blockIdx.x * 64;                 // 4 d-tiles
  int b = blockIdx.y;
  const _Float16* A = attn + (size_t)b * N_ * L_;
  const _Float16* V = vx + (size_t)b * L_ * D_;
  v8f c[4] = {};
  for (int k = 0; k < L_; k += 32) {
    v16h a = load_a16x32(A, L_, k, lane);
#pragma unroll
    for (int t = 0; t < 4; ++t) {
      v16h bf = load_b32x16(V, D_, k, d0 + 16 * t, lane);
      c[t] = wmma_f16(a, bf, c[t]);
    }
  }
  int dl = lane & 15, rb = (lane >> 4) << 3;
#pragma unroll
  for (int t = 0; t < 4; ++t)
#pragma unroll
    for (int i = 0; i < 8; ++i) {
      int n = rb + i;
      float u = c[t][i] / (attn_sum[b * N_ + n] + EPS_);
      u_h[((size_t)b * N_ + n) * D_ + d0 + 16 * t + dl] = (_Float16)u;
    }
}

// g = silu(u@w1T) * (u@w2T); 2 h-tiles x {w1,w2} share the A-fragment.
__global__ void k_swiglu1(const _Float16* __restrict__ u_h,
                          const _Float16* __restrict__ w1T,
                          const _Float16* __restrict__ w2T,
                          _Float16* __restrict__ g_h) {
  int lane = threadIdx.x;
  int h0 = blockIdx.x * 32;                 // 2 h-tiles
  int b = blockIdx.y;
  const _Float16* A = u_h + (size_t)b * N_ * D_;
  v8f c1[2] = {};
  v8f c2[2] = {};
  for (int k = 0; k < D_; k += 32) {
    v16h a = load_a16x32(A, D_, k, lane);
#pragma unroll
    for (int t = 0; t < 2; ++t) {
      v16h b1 = load_b32x16(w1T, H_, k, h0 + 16 * t, lane);
      v16h b2 = load_b32x16(w2T, H_, k, h0 + 16 * t, lane);
      c1[t] = wmma_f16(a, b1, c1[t]);
      c2[t] = wmma_f16(a, b2, c2[t]);
    }
  }
  int dl = lane & 15, rb = (lane >> 4) << 3;
#pragma unroll
  for (int t = 0; t < 2; ++t)
#pragma unroll
    for (int i = 0; i < 8; ++i) {
      float x1 = c1[t][i];
      float g = x1 * sigmoidf_(x1) * c2[t][i];
      g_h[((size_t)b * N_ + rb + i) * H_ + h0 + 16 * t + dl] = (_Float16)g;
    }
}

// u2 = g @ w3T; 4 column-tiles share the A-fragment.
__global__ void k_swiglu2(const _Float16* __restrict__ g_h,
                          const _Float16* __restrict__ w3T,
                          _Float16* __restrict__ u2_h) {
  int lane = threadIdx.x;
  int d0 = blockIdx.x * 64;                 // 4 d-tiles
  int b = blockIdx.y;
  const _Float16* A = g_h + (size_t)b * N_ * H_;
  v8f c[4] = {};
  for (int k = 0; k < H_; k += 32) {
    v16h a = load_a16x32(A, H_, k, lane);
#pragma unroll
    for (int t = 0; t < 4; ++t) {
      v16h bf = load_b32x16(w3T, D_, k, d0 + 16 * t, lane);
      c[t] = wmma_f16(a, bf, c[t]);
    }
  }
  int dl = lane & 15, rb = (lane >> 4) << 3;
#pragma unroll
  for (int t = 0; t < 4; ++t)
#pragma unroll
    for (int i = 0; i < 8; ++i)
      u2_h[((size_t)b * N_ + rb + i) * D_ + d0 + 16 * t + dl] =
          (_Float16)c[t][i];
}

// gi = u2 @ W_ihT, gh = slots @ W_hhT fused; 2 j-tiles per wave.
__global__ void k_gru_gemm(const _Float16* __restrict__ u2_h,
                           const _Float16* __restrict__ slots_h,
                           const _Float16* __restrict__ wihT,
                           const _Float16* __restrict__ whhT,
                           float* __restrict__ gi, float* __restrict__ gh) {
  int lane = threadIdx.x;
  int j0 = blockIdx.x * 32;                 // 2 tiles over 3D
  int b = blockIdx.y;
  const _Float16* A1 = u2_h + (size_t)b * N_ * D_;
  const _Float16* A2 = slots_h + (size_t)b * N_ * D_;
  v8f ci[2] = {};
  v8f ch[2] = {};
  for (int k = 0; k < D_; k += 32) {
    v16h a1 = load_a16x32(A1, D_, k, lane);
    v16h a2 = load_a16x32(A2, D_, k, lane);
#pragma unroll
    for (int t = 0; t < 2; ++t) {
      v16h b1 = load_b32x16(wihT, 3 * D_, k, j0 + 16 * t, lane);
      v16h b2 = load_b32x16(whhT, 3 * D_, k, j0 + 16 * t, lane);
      ci[t] = wmma_f16(a1, b1, ci[t]);
      ch[t] = wmma_f16(a2, b2, ch[t]);
    }
  }
  int dl = lane & 15, rb = (lane >> 4) << 3;
#pragma unroll
  for (int t = 0; t < 2; ++t)
#pragma unroll
    for (int i = 0; i < 8; ++i) {
      size_t o = ((size_t)b * N_ + rb + i) * (3 * D_) + j0 + 16 * t + dl;
      gi[o] = ci[t][i];
      gh[o] = ch[t][i];
    }
}

// GRU gating (torch ordering r,z,n; b_hh_n multiplied by r like the reference)
__global__ void k_gru_elem(const float* __restrict__ gi,
                           const float* __restrict__ gh,
                           const float* __restrict__ b_ih,
                           const float* __restrict__ b_hh,
                           float* __restrict__ slots) {
  int idx = blockIdx.x * blockDim.x + threadIdx.x;
  if (idx >= B_ * N_ * D_) return;
  int bn = idx / D_, d = idx % D_;
  size_t base = (size_t)bn * 3 * D_;
  float r = sigmoidf_(gi[base + d] + b_ih[d] + gh[base + d] + b_hh[d]);
  float z = sigmoidf_(gi[base + D_ + d] + b_ih[D_ + d] +
                      gh[base + D_ + d] + b_hh[D_ + d]);
  float ng = tanhf(gi[base + 2 * D_ + d] + b_ih[2 * D_ + d] +
                   r * (gh[base + 2 * D_ + d] + b_hh[2 * D_ + d]));
  float h = slots[idx];
  slots[idx] = (1.0f - z) * ng + z * h;
}

// ---------------------------------------------------------------------------

extern "C" void kernel_launch(void* const* d_in, const int* in_sizes, int n_in,
                              void* d_out, int out_size, void* d_ws,
                              size_t ws_size, hipStream_t stream) {
  (void)in_sizes; (void)n_in; (void)out_size; (void)ws_size;
  const float* enc     = (const float*)d_in[0];
  const float* noise   = (const float*)d_in[1];
  const float* mu      = (const float*)d_in[2];
  const float* sigma   = (const float*)d_in[3];
  const float* g_in    = (const float*)d_in[4];
  const float* g_slots = (const float*)d_in[5];
  const float* Wq      = (const float*)d_in[6];
  const float* Wk      = (const float*)d_in[7];
  const float* Wv      = (const float*)d_in[8];
  const float* w1      = (const float*)d_in[9];
  const float* w2      = (const float*)d_in[10];
  const float* w3      = (const float*)d_in[11];
  const float* Wih     = (const float*)d_in[12];
  const float* Whh     = (const float*)d_in[13];
  const float* b_ih    = (const float*)d_in[14];
  const float* b_hh    = (const float*)d_in[15];

  char* ws = (char*)d_ws;
  size_t off = 0;
  auto alloc = [&](size_t bytes) -> void* {
    void* p = ws + off;
    off += (bytes + 255) & ~(size_t)255;
    return p;
  };

  _Float16* x_h     = (_Float16*)alloc((size_t)B_ * L_ * D_ * 2);
  _Float16* k_h     = (_Float16*)alloc((size_t)B_ * L_ * D_ * 2);
  _Float16* v_h     = (_Float16*)alloc((size_t)B_ * L_ * D_ * 2);
  _Float16* WqT     = (_Float16*)alloc((size_t)D_ * D_ * 2);
  _Float16* WkT     = (_Float16*)alloc((size_t)D_ * D_ * 2);
  _Float16* WvT     = (_Float16*)alloc((size_t)D_ * D_ * 2);
  _Float16* w1T     = (_Float16*)alloc((size_t)D_ * H_ * 2);
  _Float16* w2T     = (_Float16*)alloc((size_t)D_ * H_ * 2);
  _Float16* w3T     = (_Float16*)alloc((size_t)H_ * D_ * 2);
  _Float16* WihT    = (_Float16*)alloc((size_t)D_ * 3 * D_ * 2);
  _Float16* WhhT    = (_Float16*)alloc((size_t)D_ * 3 * D_ * 2);
  float*    slots   = (float*)   alloc((size_t)B_ * N_ * D_ * 4);
  _Float16* s_h     = (_Float16*)alloc((size_t)B_ * N_ * D_ * 2);
  _Float16* slots_h = (_Float16*)alloc((size_t)B_ * N_ * D_ * 2);
  _Float16* qT      = (_Float16*)alloc((size_t)B_ * D_ * N_ * 2);
  _Float16* attn_h  = (_Float16*)alloc((size_t)B_ * N_ * L_ * 2);
  float*    attn_sum= (float*)   alloc((size_t)B_ * N_ * 4);
  _Float16* u_h     = (_Float16*)alloc((size_t)B_ * N_ * D_ * 2);
  _Float16* g_h     = (_Float16*)alloc((size_t)B_ * N_ * H_ * 2);
  _Float16* u2_h    = (_Float16*)alloc((size_t)B_ * N_ * D_ * 2);
  float*    gi_f    = (float*)   alloc((size_t)B_ * N_ * 3 * D_ * 4);
  float*    gh_f    = (float*)   alloc((size_t)B_ * N_ * 3 * D_ * 4);

  auto T = [&](const float* src, _Float16* dst, int R, int C) {
    int n = R * C;
    k_transpose_f16<<<(n + 255) / 256, 256, 0, stream>>>(src, dst, R, C);
  };
  T(Wq, WqT, D_, D_);
  T(Wk, WkT, D_, D_);
  T(Wv, WvT, D_, D_);
  T(w1, w1T, H_, D_);
  T(w2, w2T, H_, D_);
  T(w3, w3T, D_, H_);
  T(Wih, WihT, 3 * D_, D_);
  T(Whh, WhhT, 3 * D_, D_);

  k_slots_init<<<(B_ * N_ * D_ + 255) / 256, 256, 0, stream>>>(noise, mu, sigma,
                                                               slots);
  k_rmsnorm_f16<<<B_ * L_, 256, 0, stream>>>(enc, g_in, x_h);
  k_kv_gemm<<<dim3(D_ / 16, (B_ * L_) / 64), 32, 0, stream>>>(x_h, WkT, WvT,
                                                              k_h, v_h);

  for (int it = 0; it < NIT_; ++it) {
    k_slot_prep<<<B_ * N_, 256, 0, stream>>>(slots, g_slots, s_h, slots_h,
                                             attn_sum);
    k_q_proj<<<dim3(D_ / 64, B_), 32, 0, stream>>>(s_h, WqT, qT);
    k_attn<<<dim3(L_ / 64, B_), 32, 0, stream>>>(k_h, qT, attn_h, attn_sum);
    k_updates<<<dim3(D_ / 64, B_), 32, 0, stream>>>(attn_h, v_h, attn_sum, u_h);
    k_swiglu1<<<dim3(H_ / 32, B_), 32, 0, stream>>>(u_h, w1T, w2T, g_h);
    k_swiglu2<<<dim3(D_ / 64, B_), 32, 0, stream>>>(g_h, w3T, u2_h);
    k_gru_gemm<<<dim3(3 * D_ / 32, B_), 32, 0, stream>>>(u2_h, slots_h, WihT,
                                                         WhhT, gi_f, gh_f);
    k_gru_elem<<<(B_ * N_ * D_ + 255) / 256, 256, 0, stream>>>(gi_f, gh_f,
                                                               b_ih, b_hh,
                                                               slots);
  }

  hipMemcpyAsync(d_out, slots, (size_t)B_ * N_ * D_ * sizeof(float),
                 hipMemcpyDeviceToDevice, stream);
}
